// LSTMAutoencoder_10307921510453
// MI455X (gfx1250) — matrix operand order repaired and checked by reading, not compiled, not run on hardware
//
#include <hip/hip_runtime.h>
#include <stdint.h>

typedef __attribute__((ext_vector_type(16))) __bf16 v16bf;
typedef __attribute__((ext_vector_type(8)))  float  v8f;

#define Bsz  512
#define Tsz  128
#define DIN  128
#define DHID 256

__device__ __forceinline__ __bf16 f2bf(float f) {
    union { float f; unsigned u; } x; x.f = f;
    unsigned r = x.u + 0x7FFFu + ((x.u >> 16) & 1u);   // RNE
    unsigned short hs = (unsigned short)(r >> 16);
    __bf16 b; __builtin_memcpy(&b, &hs, 2);
    return b;
}
__device__ __forceinline__ float sigm(float x) { return 1.f / (1.f + __expf(-x)); }

// CDNA5 async global->LDS DMA (ASYNCcnt path), 16B per lane.
// vdst = 32-bit LDS byte address (low 32 bits of the generic pointer).
__device__ __forceinline__ void async_ld16(unsigned lds_off, unsigned long long gaddr) {
    asm volatile("global_load_async_to_lds_b128 %0, %1, off"
                 :: "v"(lds_off), "v"(gaddr) : "memory");
}
__device__ __forceinline__ void wait_async0() {
    asm volatile("s_wait_asynccnt 0x0" ::: "memory");
}
__device__ __forceinline__ unsigned lds_addr(const void* p) {
    return (unsigned)(uintptr_t)p;   // flat LDS addr = {aperture, offset[31:0]}
}

// ---------------------------------------------------------------------------
// Fused LSTM step:  g = [x_t | h_prev] @ Wcat^T + bias ; gate math in regs.
// Block = 128 threads (4 waves). Block tile: 64 batch rows x 16 hidden units,
// each wave owns a 16x16 (m,j) tile across ALL FOUR gates (4 accumulators).
// Double-buffered LDS; staging via async DMA overlapped with WMMA.
// ---------------------------------------------------------------------------
__global__ __launch_bounds__(128)
void lstm_step_kernel(const __bf16* __restrict__ xin, int xb_stride, int xt_stride, int Din,
                      const __bf16* __restrict__ Wcat, const float* __restrict__ bias,
                      int Kc, int H,
                      const __bf16* __restrict__ h_in, __bf16* __restrict__ h_out,
                      float* __restrict__ c,
                      __bf16* __restrict__ y, int y_bstride, int t)
{
    __shared__ __bf16 As[2][64 * 32];   // [buf][row 0..63][k 0..31]
    __shared__ __bf16 Bs[2][64 * 32];   // [buf][gate*16+j][k 0..31]

    const int tid  = threadIdx.x;
    const int wave = tid >> 5;
    const int lane = tid & 31;
    const int m0   = blockIdx.x * 64;
    const int j0   = blockIdx.y * 16;

    v8f acc[4];
#pragma unroll
    for (int g = 0; g < 4; ++g)
#pragma unroll
        for (int e = 0; e < 8; ++e) acc[g][e] = 0.f;

    // ---- async stage of one 64x32 A tile + 64x32 B tile into LDS buf ----
    auto stage = [&](int buf, int kc) {
#pragma unroll
        for (int it = 0; it < 2; ++it) {
            int chunk = tid + it * 128;          // 256 x 16B chunks (A)
            int row = chunk >> 2, seg = chunk & 3;
            int k = kc + seg * 8;
            const __bf16* src;
            if (k < Din) src = xin + (size_t)(m0 + row) * xb_stride + (size_t)t * xt_stride + k;
            else         src = h_in + (size_t)(m0 + row) * H + (k - Din);
            async_ld16(lds_addr(&As[buf][row * 32 + seg * 8]),
                       (unsigned long long)(uintptr_t)src);
        }
#pragma unroll
        for (int it = 0; it < 2; ++it) {
            int chunk = tid + it * 128;          // 256 x 16B chunks (B)
            int nl = chunk >> 2, seg = chunk & 3;
            int g = nl >> 4, jj = nl & 15;
            const __bf16* src = Wcat + (size_t)(g * H + j0 + jj) * Kc + kc + seg * 8;
            async_ld16(lds_addr(&Bs[buf][nl * 32 + seg * 8]),
                       (unsigned long long)(uintptr_t)src);
        }
    };

    const int nIter = Kc >> 5;
    stage(0, 0);
    wait_async0();
    __syncthreads();

    union Frag { v16bf v; uint4 q[2]; };
    for (int i = 0; i < nIter; ++i) {
        const int cur = i & 1;
        if (i + 1 < nIter) stage(cur ^ 1, (i + 1) * 32);   // DMA next tile

        // A fragment (16x32 bf16): lanes 0-15 row M with k {0..7,16..23};
        // lanes 16-31 same rows with k {8..15,24..31}.
        Frag a;
        {
            const __bf16* Ar = &As[cur][(wave * 16 + (lane & 15)) * 32];
            int kb = (lane < 16) ? 0 : 8;
            a.q[0] = *(const uint4*)(Ar + kb);
            a.q[1] = *(const uint4*)(Ar + kb + 16);
        }
        // B fragments for all four gates (32x16 bf16): lane%16 = column,
        // lanes<16 hold k {0..7,16..23}, lanes>=16 hold k {8..15,24..31}.
        Frag b[4];
        const int kbB = (lane < 16) ? 0 : 8;
#pragma unroll
        for (int g = 0; g < 4; ++g) {
            const __bf16* Br = &Bs[cur][(g * 16 + (lane & 15)) * 32 + kbB];
            b[g].q[0] = *(const uint4*)Br;
            b[g].q[1] = *(const uint4*)(Br + 16);
        }
#pragma unroll
        for (int g = 0; g < 4; ++g)
            acc[g] = __builtin_amdgcn_wmma_f32_16x16x32_bf16(
                false, a.v, false, b[g].v, (short)0, acc[g], false, false);

        wait_async0();       // this wave's DMA into the other buffer is done
        __syncthreads();     // everyone finished reading cur / writing next
    }

    // ---- fused gate epilogue, entirely in registers ----
    const int j     = j0 + (lane & 15);
    const int mbase = m0 + wave * 16 + ((lane < 16) ? 0 : 8);
#pragma unroll
    for (int r = 0; r < 8; ++r) {
        int m = mbase + r;
        float gi = acc[0][r] + bias[0 * H + j];
        float gf = acc[1][r] + bias[1 * H + j];
        float gg = acc[2][r] + bias[2 * H + j];
        float go = acc[3][r] + bias[3 * H + j];
        float cprev = c[(size_t)m * H + j];
        float cn = sigm(gf) * cprev + sigm(gi) * tanhf(gg);
        float hn = sigm(go) * tanhf(cn);
        c[(size_t)m * H + j] = cn;
        __bf16 hb = f2bf(hn);
        h_out[(size_t)m * H + j] = hb;
        if (y) y[(size_t)m * y_bstride + (size_t)t * H + j] = hb;
    }
}

// ---------------------------------------------------------------------------
// Final linear: O[m,n] = sum_k A[m,k]*W[n,k] + bias[n].  M=B*T, N=K=128.
// Block tile 64x16, one 16x16 WMMA tile per wave.
// ---------------------------------------------------------------------------
__global__ __launch_bounds__(128)
void gemm_out_kernel(const __bf16* __restrict__ A, const __bf16* __restrict__ W,
                     const float* __restrict__ bias, float* __restrict__ O)
{
    __shared__ __bf16 As[64 * 32];
    __shared__ __bf16 Bs[16 * 32];

    const int tid  = threadIdx.x;
    const int wave = tid >> 5;
    const int lane = tid & 31;
    const int m0   = blockIdx.x * 64;
    const int n0   = blockIdx.y * 16;

    v8f acc;
#pragma unroll
    for (int e = 0; e < 8; ++e) acc[e] = 0.f;

    for (int kc = 0; kc < 128; kc += 32) {
#pragma unroll
        for (int it = 0; it < 2; ++it) {
            int chunk = tid + it * 128;
            int row = chunk >> 2, seg = chunk & 3;
            *(uint4*)&As[row * 32 + seg * 8] =
                *(const uint4*)(A + (size_t)(m0 + row) * 128 + kc + seg * 8);
        }
        if (tid < 64) {
            int nl = tid >> 2, seg = tid & 3;
            *(uint4*)&Bs[nl * 32 + seg * 8] =
                *(const uint4*)(W + (size_t)(n0 + nl) * 128 + kc + seg * 8);
        }
        __syncthreads();

        union Frag { v16bf v; uint4 q[2]; };
        Frag a, b;
        {
            const __bf16* Ar = &As[(wave * 16 + (lane & 15)) * 32];
            int kb = (lane < 16) ? 0 : 8;
            a.q[0] = *(const uint4*)(Ar + kb);
            a.q[1] = *(const uint4*)(Ar + kb + 16);
        }
        {
            const __bf16* Br = &Bs[(lane & 15) * 32 + ((lane < 16) ? 0 : 8)];
            b.q[0] = *(const uint4*)Br;
            b.q[1] = *(const uint4*)(Br + 16);
        }
        acc = __builtin_amdgcn_wmma_f32_16x16x32_bf16(
            false, a.v, false, b.v, (short)0, acc, false, false);
        __syncthreads();
    }

    const int n     = n0 + (lane & 15);
    const int mbase = m0 + wave * 16 + ((lane < 16) ? 0 : 8);
#pragma unroll
    for (int r = 0; r < 8; ++r)
        O[(size_t)(mbase + r) * 128 + n] = acc[r] + bias[n];
}

// ------------------------------- prep kernels ------------------------------
__global__ void cvt_f32_bf16_kernel(const float* __restrict__ s, __bf16* __restrict__ d, int n) {
    int i = blockIdx.x * blockDim.x + threadIdx.x;
    if (i < n) d[i] = f2bf(s[i]);
}
__global__ void build_wcat_kernel(const float* __restrict__ Wih, const float* __restrict__ Whh,
                                  __bf16* __restrict__ dst, int rows, int Din, int H, int Kc) {
    int i = blockIdx.x * blockDim.x + threadIdx.x;
    if (i >= rows * Kc) return;
    int n = i / Kc, k = i - n * Kc;
    float v = (k < Din) ? Wih[(size_t)n * Din + k] : Whh[(size_t)n * H + (k - Din)];
    dst[i] = f2bf(v);
}
__global__ void bias_sum_kernel(const float* __restrict__ a, const float* __restrict__ b,
                                float* __restrict__ d, int n) {
    int i = blockIdx.x * blockDim.x + threadIdx.x;
    if (i < n) d[i] = a[i] + b[i];
}
__global__ void zero_u32_kernel(unsigned* __restrict__ p, int n) {
    int i = blockIdx.x * blockDim.x + threadIdx.x;
    if (i < n) p[i] = 0u;
}

// ------------------------------- host driver -------------------------------
extern "C" void kernel_launch(void* const* d_in, const int* in_sizes, int n_in,
                              void* d_out, int out_size, void* d_ws, size_t ws_size,
                              hipStream_t stream)
{
    const float* x        = (const float*)d_in[0];
    const float* Wih[4]   = { (const float*)d_in[1],  (const float*)d_in[5],
                              (const float*)d_in[9],  (const float*)d_in[13] };
    const float* Whh[4]   = { (const float*)d_in[2],  (const float*)d_in[6],
                              (const float*)d_in[10], (const float*)d_in[14] };
    const float* bih[4]   = { (const float*)d_in[3],  (const float*)d_in[7],
                              (const float*)d_in[11], (const float*)d_in[15] };
    const float* bhh[4]   = { (const float*)d_in[4],  (const float*)d_in[8],
                              (const float*)d_in[12], (const float*)d_in[16] };
    const float* out_W    = (const float*)d_in[17];
    const float* out_b    = (const float*)d_in[18];
    float* out            = (float*)d_out;

    uintptr_t p = (uintptr_t)d_ws;
    auto carve = [&](size_t bytes) -> void* {
        void* r = (void*)((p + 255) & ~(uintptr_t)255);
        p = ((p + 255) & ~(uintptr_t)255) + bytes;
        return r;
    };

    // big activation regions (with reuse: xbf->y2, y0->y3)
    __bf16* xbf = (__bf16*)carve((size_t)Bsz * Tsz * DIN * 2);   // enc0 input; later y2
    __bf16* y0  = (__bf16*)carve((size_t)Bsz * Tsz * DHID * 2);  // enc0 out; later y3
    __bf16* y2  = xbf;
    __bf16* y3  = y0;

    // concatenated bf16 weights [4H x (Din+H)] and summed biases
    const int Hs[4]   = { DHID, DHID, DIN, DIN };
    const int Dins[4] = { DIN,  DHID, DHID, DIN };
    __bf16* Wc[4]; float* bs[4];
    for (int l = 0; l < 4; ++l) {
        Wc[l] = (__bf16*)carve((size_t)4 * Hs[l] * (Dins[l] + Hs[l]) * 2);
        bs[l] = (float*)carve((size_t)4 * Hs[l] * 4);
    }
    __bf16* Wob = (__bf16*)carve((size_t)DIN * DIN * 2);

    // per-layer hidden/cell state (h double-buffered)
    __bf16* hA[4]; __bf16* hB[4]; float* cS[4];
    for (int l = 0; l < 4; ++l) {
        hA[l] = (__bf16*)carve((size_t)Bsz * Hs[l] * 2);
        hB[l] = (__bf16*)carve((size_t)Bsz * Hs[l] * 2);
        cS[l] = (float*)carve((size_t)Bsz * Hs[l] * 4);
    }

    // ---- prep: bf16 conversions, weight concat, bias sums ----
    {
        int n = Bsz * Tsz * DIN;
        cvt_f32_bf16_kernel<<<(n + 255) / 256, 256, 0, stream>>>(x, xbf, n);
        n = DIN * DIN;
        cvt_f32_bf16_kernel<<<(n + 255) / 256, 256, 0, stream>>>(out_W, Wob, n);
        for (int l = 0; l < 4; ++l) {
            int Kc = Dins[l] + Hs[l];
            int ne = 4 * Hs[l] * Kc;
            build_wcat_kernel<<<(ne + 255) / 256, 256, 0, stream>>>(
                Wih[l], Whh[l], Wc[l], 4 * Hs[l], Dins[l], Hs[l], Kc);
            int nb = 4 * Hs[l];
            bias_sum_kernel<<<(nb + 255) / 256, 256, 0, stream>>>(bih[l], bhh[l], bs[l], nb);
        }
    }

    auto run_layer = [&](int l, const __bf16* xin, int xb, int xt,
                         __bf16* y) -> __bf16* {
        int H = Hs[l], Din = Dins[l], Kc = Din + H;
        int nh = Bsz * H / 2, nc = Bsz * H;
        zero_u32_kernel<<<(nh + 255) / 256, 256, 0, stream>>>((unsigned*)hA[l], nh);
        zero_u32_kernel<<<(nh + 255) / 256, 256, 0, stream>>>((unsigned*)hB[l], nh);
        zero_u32_kernel<<<(nc + 255) / 256, 256, 0, stream>>>((unsigned*)cS[l], nc);
        dim3 grid(Bsz / 64, H / 16);
        for (int t = 0; t < Tsz; ++t) {
            const __bf16* hin = (t & 1) ? hB[l] : hA[l];
            __bf16*       hout = (t & 1) ? hA[l] : hB[l];
            lstm_step_kernel<<<grid, 128, 0, stream>>>(
                xin, xb, xt, Din, Wc[l], bs[l], Kc, H,
                hin, hout, cS[l], y, y ? Tsz * H : 0, t);
        }
        return ((Tsz - 1) & 1) ? hA[l] : hB[l];   // h after final step
    };

    // encoder
    run_layer(0, xbf, Tsz * DIN, DIN, y0);
    __bf16* hT = run_layer(1, y0, Tsz * DHID, DHID, nullptr);
    // decoder (repeat-vector: t-stride 0 broadcasts hT over time)
    run_layer(2, hT, DHID, 0, y2);
    run_layer(3, y2, Tsz * DIN, DIN, y3);
    // final linear -> f32 output [B*T, 128]
    gemm_out_kernel<<<dim3(Bsz * Tsz / 64, DIN / 16), 128, 0, stream>>>(y3, Wob, out_b, out);
    (void)in_sizes; (void)n_in; (void)out_size; (void)ws_size;
}
